// RAFT3D_32177894982190
// MI455X (gfx1250) — matrix operand order repaired and checked by reading, not compiled, not run on hardware
//
#include <hip/hip_runtime.h>
#include <hip/hip_bf16.h>

typedef _Float16 v8h  __attribute__((ext_vector_type(8)));
typedef _Float16 v16h __attribute__((ext_vector_type(16)));
typedef float    v8f  __attribute__((ext_vector_type(8)));

// exact payload type from the builtin's signature: int4 (16 bytes)
typedef int v4i_vs __attribute__((vector_size(4 * sizeof(int))));
typedef __attribute__((address_space(1))) v4i_vs as1_v4i;  // global
typedef __attribute__((address_space(3))) v4i_vs as3_v4i;  // LDS

#if defined(__has_builtin)
#if __has_builtin(__builtin_amdgcn_global_load_async_to_lds_b128)
#define HAVE_ASYNC_LDS 1
#endif
#endif
#ifndef HAVE_ASYNC_LDS
#define HAVE_ASYNC_LDS 0
#endif

constexpr int NB = 8, H = 60, W = 80, HW = H * W;   // 4800
constexpr int NPIX = NB * HW;                       // 38400 = 300 * 128
constexpr int HD = 128;

enum { EP_NONE = 0, EP_RELU = 1, EP_SIG = 2 };

// ---------------------------------------------------------------------------
// Implicit-GEMM convolution via V_WMMA_F32_16X16X32_F16.
//   A : f16 weights  [M][KS*KS*Cpad]   (O-(ky,kx)-Ipad, zero padded)
//   X : f16 activations NHWC-padded [NB*H*W][Cpad]  (Cpad % 32 == 0)
//   out16: NHWC-padded [pixel][Mpad];  out32: NCHW [NB][M][H][W]
// Block 256 threads = 8 waves; tile 64(M) x 128(pixels); waves 4(M) x 2(N),
// each wave owns a 16x64 strip -> 4 f32 accumulators, 4 WMMA per K-step.
// Double-buffered LDS: K-step k+32 is staged (async global->LDS copies on
// CDNA5, ASYNCcnt-tracked) while the WMMAs of step k run; one barrier/step.
// OOB lanes (conv halo / M tail) zero their LDS slot with a DS store instead.
// ---------------------------------------------------------------------------
template <int KS, int PAD>
__global__ __launch_bounds__(256) void conv_wmma(
    const _Float16* __restrict__ A, const _Float16* __restrict__ X,
    const float* __restrict__ bias, _Float16* __restrict__ out16,
    float* __restrict__ out32, int M, int Cpad, int Mpad, int ep) {
  const int K = Cpad * KS * KS;
  const int n0 = blockIdx.x * 128;
  const int m0 = blockIdx.y * 64;
  const int tid = threadIdx.x;
  const int wave = tid >> 5;
  const int lane = tid & 31;
  const int mw = (wave & 3) * 16;   // wave's M offset in tile
  const int nw = (wave >> 2) * 64;  // wave's N offset in tile (0 or 64)

  __shared__ _Float16 As[2][64][40];   // rows padded to 80B (bank spread)
  __shared__ _Float16 Bs[2][128][40];

  const int arow = tid >> 2;        // 0..63
  const int kl = (tid & 3) * 8;     // 0,8,16,24

  // two pixels per thread for the B tile (128-pixel tile / 64 pixel-slots)
  int pb[2], ph[2], pw[2];
  bool pok[2];
#pragma unroll
  for (int r = 0; r < 2; ++r) {
    const int n = n0 + arow + 64 * r;
    pok[r] = (n < NPIX);
    const int b = n / HW, p = n - b * HW;
    pb[r] = b; ph[r] = p / W; pw[r] = p - (p / W) * W;
  }

  // (ky,kx,channel-offset) of the K-step currently being staged
  int ky = 0, kx = 0, coff = 0;

  // one 16-byte tile-chunk copy: async global->LDS on CDNA5, sync fallback
  auto cp16 = [&](_Float16* ldst, const _Float16* gsrc) {
#if HAVE_ASYNC_LDS
    __builtin_amdgcn_global_load_async_to_lds_b128((as1_v4i*)gsrc,
                                                   (as3_v4i*)ldst, 0, 0);
#else
    *(v8h*)ldst = *(const v8h*)gsrc;
#endif
  };
  auto zero16 = [&](_Float16* ldst) {
    v8h z = {};
    *(v8h*)ldst = z;
  };
  auto stage = [&](int sbuf, int k0) {
    // A tile row-chunk
    const int gm = m0 + arow;
    if (gm < M)
      cp16(&As[sbuf][arow][kl], A + (size_t)gm * K + k0 + kl);
    else
      zero16(&As[sbuf][arow][kl]);
    // B tile pixel-chunks (im2col with halo zero-fill)
#pragma unroll
    for (int r = 0; r < 2; ++r) {
      const int y = ph[r] + ky - PAD;
      const int x = pw[r] + kx - PAD;
      if (pok[r] && (unsigned)y < (unsigned)H && (unsigned)x < (unsigned)W)
        cp16(&Bs[sbuf][arow + 64 * r][kl],
             X + ((size_t)(pb[r] * H + y) * W + x) * Cpad + coff + kl);
      else
        zero16(&Bs[sbuf][arow + 64 * r][kl]);
    }
  };
  auto wait_stage = [&]() {
#if HAVE_ASYNC_LDS
#if __has_builtin(__builtin_amdgcn_s_wait_asynccnt)
    __builtin_amdgcn_s_wait_asynccnt(0);
#else
    asm volatile("s_wait_asynccnt 0" ::: "memory");
#endif
#endif
  };

  v8f acc[4] = {};

  stage(0, 0);  // prologue: stage K-step 0 into buffer 0

  const int r16 = lane & 15;
  const int ka = (lane < 16) ? 0 : 16;   // A frag: K {0-15}/{16-31}
  const int kb = (lane < 16) ? 0 : 16;   // B frag: K {0-15}/{16-31}

  int cur = 0;
  int k0 = 32;
  for (;;) {
    wait_stage();
    __syncthreads();                     // buf[cur] ready for all waves
    const bool more = (k0 < K);
    if (more) {
      // advance (ky,kx,coff) to k0 (exact since Cpad % 32 == 0)
      coff += 32;
      if (coff == Cpad) {
        coff = 0;
        if (KS > 1) { ++kx; if (kx == KS) { kx = 0; ++ky; } }
      }
      stage(cur ^ 1, k0);                // overlap copy with WMMAs below
    }

    // fragments + 4 WMMAs from buf[cur]
    v16h af;
    {
      v8h lo = *(const v8h*)&As[cur][mw + r16][ka];
      v8h hi = *(const v8h*)&As[cur][mw + r16][ka + 8];
#pragma unroll
      for (int j = 0; j < 8; ++j) { af[j] = lo[j]; af[8 + j] = hi[j]; }
    }
#pragma unroll
    for (int t = 0; t < 4; ++t) {
      v16h bf;
      v8h lo = *(const v8h*)&Bs[cur][nw + t * 16 + r16][kb];
      v8h hi = *(const v8h*)&Bs[cur][nw + t * 16 + r16][kb + 8];
#pragma unroll
      for (int j = 0; j < 8; ++j) { bf[j] = lo[j]; bf[8 + j] = hi[j]; }
      acc[t] = __builtin_amdgcn_wmma_f32_16x16x32_f16(false, af, false, bf,
                                                      (short)0, acc[t], false,
                                                      false);
    }

    if (!more) break;
    cur ^= 1;
    k0 += 32;
  }

  // ---- fused epilogue: bias + activation ----------------------------------
  const int mbase = m0 + mw + ((lane < 16) ? 0 : 8);
#pragma unroll
  for (int v = 0; v < 8; ++v) {
    const int m = mbase + v;
    if (m >= M) continue;
    const float bv = bias[m];
#pragma unroll
    for (int t = 0; t < 4; ++t) {
      const int nn = n0 + nw + t * 16 + r16;
      if (nn >= NPIX) continue;
      float val = acc[t][v] + bv;
      if (ep == EP_RELU) val = fmaxf(val, 0.0f);
      else if (ep == EP_SIG) val = 1.0f / (1.0f + __expf(-val));
      if (out16) out16[(size_t)nn * Mpad + m] = (_Float16)val;
      if (out32) {
        const int bb = nn / HW, pp = nn - bb * HW;
        out32[((size_t)bb * M + m) * HW + pp] = val;
      }
    }
  }
}

// ---------------------------------------------------------------------------
// Pack kernels
// ---------------------------------------------------------------------------
// OIHW f32 -> [M][KK][Cpad] f16 (zero padded channels)
__global__ void k_pack_w(const float* __restrict__ s, _Float16* __restrict__ d,
                         int M, int Cin, int Cpad, int KK) {
  int i = blockIdx.x * blockDim.x + threadIdx.x;
  const int n = M * KK * Cpad;
  if (i >= n) return;
  const int o = i / (KK * Cpad);
  const int r = i - o * (KK * Cpad);
  const int seg = r / Cpad;
  const int c = r - seg * Cpad;
  d[i] = (c < Cin) ? (_Float16)s[((size_t)o * Cin + c) * KK + seg]
                   : (_Float16)0.0f;
}

// NCHW f32 -> NHWC-padded f16
__global__ void k_pack_x(const float* __restrict__ s, _Float16* __restrict__ d,
                         int C, int Cpad) {
  int i = blockIdx.x * blockDim.x + threadIdx.x;
  if (i >= NPIX * Cpad) return;
  const int n = i / Cpad, c = i - n * Cpad;
  const int b = n / HW, p = n - b * HW;
  d[i] = (c < C) ? (_Float16)s[((size_t)b * C + c) * HW + p] : (_Float16)0.0f;
}

// motion = clip(cat(flow, 10*dz, 10*twist), ±50) -> NHWC-padded f16 (9 -> 32)
__global__ void k_motion(const float* __restrict__ flow,
                         const float* __restrict__ dz,
                         const float* __restrict__ twist,
                         _Float16* __restrict__ mot) {
  int i = blockIdx.x * blockDim.x + threadIdx.x;  // pixel index
  if (i >= NPIX) return;
  float m[9];
  m[0] = flow[(size_t)i * 2 + 0];
  m[1] = flow[(size_t)i * 2 + 1];
  m[2] = 10.0f * dz[i];
#pragma unroll
  for (int c = 0; c < 6; ++c) m[3 + c] = 10.0f * twist[(size_t)i * 6 + c];
#pragma unroll
  for (int c = 0; c < 9; ++c) {
    float v = fminf(fmaxf(m[c], -50.0f), 50.0f);
    mot[(size_t)i * 32 + c] = (_Float16)v;
  }
#pragma unroll
  for (int c = 9; c < 32; ++c) mot[(size_t)i * 32 + c] = (_Float16)0.0f;
}

// ---------------------------------------------------------------------------
// GRU elementwise (channel-minor over NHWC tensors)
// ---------------------------------------------------------------------------
__global__ void k_gru1(const _Float16* __restrict__ gz,
                       const _Float16* __restrict__ gr,
                       const float* __restrict__ inp,     // NCHW f32 (384 ch)
                       const _Float16* __restrict__ cor,  // NHWC [pix][384]
                       const _Float16* __restrict__ mot,  // NHWC [pix][384]
                       const float* __restrict__ net,     // NCHW f32 (128 ch)
                       _Float16* __restrict__ zbuf,       // NHWC [pix][128]
                       _Float16* __restrict__ rnet) {     // NHWC [pix][128]
  int i = blockIdx.x * blockDim.x + threadIdx.x;
  if (i >= NPIX * HD) return;
  const int n = i / HD, c = i - n * HD;
  const int b = n / HW, p = n - b * HW;
  const size_t j0 = (size_t)n * 384 + c;
  const size_t j1 = j0 + HD;
  const size_t i0 = ((size_t)b * (3 * HD) + c) * HW + p;
  const size_t i1 = ((size_t)b * (3 * HD) + HD + c) * HW + p;
  const float netv = net[((size_t)b * HD + c) * HW + p];
  float z = (float)gz[i] + inp[i0] + (float)cor[j0] + (float)mot[j0];
  float r = (float)gr[i] + inp[i1] + (float)cor[j1] + (float)mot[j1];
  z = 1.0f / (1.0f + __expf(-z));
  r = 1.0f / (1.0f + __expf(-r));
  zbuf[i] = (_Float16)z;
  rnet[i] = (_Float16)(r * netv);
}

__global__ void k_gru2(const _Float16* __restrict__ gq,
                       const float* __restrict__ inp,
                       const _Float16* __restrict__ cor,
                       const _Float16* __restrict__ mot,
                       const float* __restrict__ net,
                       const _Float16* __restrict__ zbuf,
                       float* __restrict__ out_net,       // NCHW f32 (d_out)
                       _Float16* __restrict__ net16) {    // NHWC [pix][128]
  int i = blockIdx.x * blockDim.x + threadIdx.x;
  if (i >= NPIX * HD) return;
  const int n = i / HD, c = i - n * HD;
  const int b = n / HW, p = n - b * HW;
  const size_t j2 = (size_t)n * 384 + 2 * HD + c;
  const size_t i2 = ((size_t)b * (3 * HD) + 2 * HD + c) * HW + p;
  const size_t inet = ((size_t)b * HD + c) * HW + p;
  float q = (float)gq[i] + inp[i2] + (float)cor[j2] + (float)mot[j2];
  q = tanhf(q);
  const float z = (float)zbuf[i];
  const float nn = (1.0f - z) * net[inet] + z * q;
  out_net[inet] = nn;
  net16[i] = (_Float16)nn;
}

// ---------------------------------------------------------------------------
extern "C" void kernel_launch(void* const* d_in, const int* in_sizes, int n_in,
                              void* d_out, int out_size, void* d_ws,
                              size_t ws_size, hipStream_t stream) {
  (void)in_sizes; (void)n_in; (void)out_size; (void)ws_size;

  const float* net  = (const float*)d_in[0];
  const float* inp  = (const float*)d_in[1];
  const float* corr = (const float*)d_in[2];
  const float* flow = (const float*)d_in[3];
  const float* twst = (const float*)d_in[4];
  const float* dz   = (const float*)d_in[5];
  // params flattened in jax-pytree (sorted-key) order
  const float* ae1_b    = (const float*)d_in[6];
  const float* ae1_w    = (const float*)d_in[7];
  const float* ae2_b    = (const float*)d_in[8];
  const float* ae2_w    = (const float*)d_in[9];
  const float* convq_b  = (const float*)d_in[10];
  const float* convq_w  = (const float*)d_in[11];
  const float* convr_b  = (const float*)d_in[12];
  const float* convr_w  = (const float*)d_in[13];
  const float* convz_b  = (const float*)d_in[14];
  const float* convz_w  = (const float*)d_in[15];
  const float* corr1_b  = (const float*)d_in[16];
  const float* corr1_w  = (const float*)d_in[17];
  const float* corr2_b  = (const float*)d_in[18];
  const float* corr2_w  = (const float*)d_in[19];
  const float* corr3_b  = (const float*)d_in[20];
  const float* corr3_w  = (const float*)d_in[21];
  const float* delta1_b = (const float*)d_in[22];
  const float* delta1_w = (const float*)d_in[23];
  const float* delta2_b = (const float*)d_in[24];
  const float* delta2_w = (const float*)d_in[25];
  const float* flow1_b  = (const float*)d_in[26];
  const float* flow1_w  = (const float*)d_in[27];
  const float* flow2_b  = (const float*)d_in[28];
  const float* flow2_w  = (const float*)d_in[29];
  const float* mask1_b  = (const float*)d_in[30];
  const float* mask1_w  = (const float*)d_in[31];
  const float* mask2_b  = (const float*)d_in[32];
  const float* mask2_w  = (const float*)d_in[33];
  const float* wt1_b    = (const float*)d_in[34];
  const float* wt1_w    = (const float*)d_in[35];
  const float* wt2_b    = (const float*)d_in[36];
  const float* wt2_w    = (const float*)d_in[37];

  // output tuple (net, mask, ae, delta, weight) concatenated flat (NCHW f32)
  float* out_net   = (float*)d_out;
  float* out_mask  = out_net + (size_t)NB * HD * HW;
  float* out_ae    = out_mask + (size_t)NB * 576 * HW;
  float* out_delta = out_ae + (size_t)NB * 32 * HW;
  float* out_wt    = out_delta + (size_t)NB * 3 * HW;

  // workspace bump allocator (256B aligned)
  char* wsb = (char*)d_ws;
  size_t off = 0;
  auto a16 = [&](size_t halves) -> _Float16* {
    _Float16* p = (_Float16*)(wsb + off);
    off += (halves * sizeof(_Float16) + 255) & ~(size_t)255;
    return p;
  };

  // f16 packed weights: [M][KK*Cpad]
  _Float16* w_ae1    = a16((size_t)256 * 9 * 128);
  _Float16* w_ae2    = a16((size_t)32 * 256);
  _Float16* w_convq  = a16((size_t)128 * 9 * 128);
  _Float16* w_convr  = a16((size_t)128 * 9 * 128);
  _Float16* w_convz  = a16((size_t)128 * 9 * 128);
  _Float16* w_corr1  = a16((size_t)256 * 9 * 224);
  _Float16* w_corr2  = a16((size_t)256 * 9 * 256);
  _Float16* w_corr3  = a16((size_t)384 * 256);
  _Float16* w_delta1 = a16((size_t)256 * 9 * 128);
  _Float16* w_delta2 = a16((size_t)3 * 256);
  _Float16* w_flow1  = a16((size_t)128 * 49 * 32);
  _Float16* w_flow2  = a16((size_t)384 * 128);
  _Float16* w_mask1  = a16((size_t)256 * 9 * 128);
  _Float16* w_mask2  = a16((size_t)576 * 256);
  _Float16* w_wt1    = a16((size_t)256 * 9 * 128);
  _Float16* w_wt2    = a16((size_t)3 * 256);

  // f16 activations, NHWC-padded [NPIX][Cpad]
  _Float16* net16  = a16((size_t)NPIX * 128);
  _Float16* corr16 = a16((size_t)NPIX * 224);
  _Float16* mot9   = a16((size_t)NPIX * 32);
  _Float16* hid128 = a16((size_t)NPIX * 128);
  _Float16* mot384 = a16((size_t)NPIX * 384);
  _Float16* ch256a = a16((size_t)NPIX * 256);
  _Float16* ch256b = a16((size_t)NPIX * 256);
  _Float16* cor384 = a16((size_t)NPIX * 384);
  _Float16* gz16   = a16((size_t)NPIX * 128);
  _Float16* gr16   = a16((size_t)NPIX * 128);
  _Float16* gq16   = a16((size_t)NPIX * 128);
  _Float16* zbuf   = a16((size_t)NPIX * 128);
  _Float16* rnet16 = a16((size_t)NPIX * 128);

  auto packw = [&](const float* s, _Float16* d, int M, int Cin, int Cpad,
                   int KK) {
    const int n = M * KK * Cpad;
    hipLaunchKernelGGL(k_pack_w, dim3((n + 255) / 256), dim3(256), 0, stream,
                       s, d, M, Cin, Cpad, KK);
  };
  packw(ae1_w, w_ae1, 256, 128, 128, 9);
  packw(ae2_w, w_ae2, 32, 256, 256, 1);
  packw(convq_w, w_convq, 128, 128, 128, 9);
  packw(convr_w, w_convr, 128, 128, 128, 9);
  packw(convz_w, w_convz, 128, 128, 128, 9);
  packw(corr1_w, w_corr1, 256, 196, 224, 9);
  packw(corr2_w, w_corr2, 256, 256, 256, 9);
  packw(corr3_w, w_corr3, 384, 256, 256, 1);
  packw(delta1_w, w_delta1, 256, 128, 128, 9);
  packw(delta2_w, w_delta2, 3, 256, 256, 1);
  packw(flow1_w, w_flow1, 128, 9, 32, 49);
  packw(flow2_w, w_flow2, 384, 128, 128, 1);
  packw(mask1_w, w_mask1, 256, 128, 128, 9);
  packw(mask2_w, w_mask2, 576, 256, 256, 1);
  packw(wt1_w, w_wt1, 256, 128, 128, 9);
  packw(wt2_w, w_wt2, 3, 256, 256, 1);

  {
    int n = NPIX * 128;
    hipLaunchKernelGGL(k_pack_x, dim3((n + 255) / 256), dim3(256), 0, stream,
                       net, net16, 128, 128);
    n = NPIX * 224;
    hipLaunchKernelGGL(k_pack_x, dim3((n + 255) / 256), dim3(256), 0, stream,
                       corr, corr16, 196, 224);
    hipLaunchKernelGGL(k_motion, dim3((NPIX + 255) / 256), dim3(256), 0,
                       stream, flow, dz, twst, mot9);
  }

  const dim3 blk(256);
  auto conv1 = [&](const _Float16* A, const _Float16* X, const float* bb,
                   int M, int Cpad, _Float16* o16, int Mpad, float* o32,
                   int ep) {
    hipLaunchKernelGGL((conv_wmma<1, 0>), dim3(NPIX / 128, (M + 63) / 64), blk,
                       0, stream, A, X, bb, o16, o32, M, Cpad, Mpad, ep);
  };
  auto conv3 = [&](const _Float16* A, const _Float16* X, const float* bb,
                   int M, int Cpad, _Float16* o16, int Mpad, float* o32,
                   int ep) {
    hipLaunchKernelGGL((conv_wmma<3, 1>), dim3(NPIX / 128, (M + 63) / 64), blk,
                       0, stream, A, X, bb, o16, o32, M, Cpad, Mpad, ep);
  };
  auto conv7 = [&](const _Float16* A, const _Float16* X, const float* bb,
                   int M, int Cpad, _Float16* o16, int Mpad, float* o32,
                   int ep) {
    hipLaunchKernelGGL((conv_wmma<7, 3>), dim3(NPIX / 128, (M + 63) / 64), blk,
                       0, stream, A, X, bb, o16, o32, M, Cpad, Mpad, ep);
  };

  // motion encoder
  conv7(w_flow1, mot9, flow1_b, 128, 32, hid128, 128, nullptr, EP_RELU);
  conv1(w_flow2, hid128, flow2_b, 384, 128, mot384, 384, nullptr, EP_NONE);
  // correlation encoder
  conv3(w_corr1, corr16, corr1_b, 256, 224, ch256a, 256, nullptr, EP_RELU);
  conv3(w_corr2, ch256a, corr2_b, 256, 256, ch256b, 256, nullptr, EP_RELU);
  conv1(w_corr3, ch256b, corr3_b, 384, 256, cor384, 384, nullptr, EP_NONE);
  // ConvGRU
  conv3(w_convz, net16, convz_b, 128, 128, gz16, 128, nullptr, EP_NONE);
  conv3(w_convr, net16, convr_b, 128, 128, gr16, 128, nullptr, EP_NONE);
  hipLaunchKernelGGL(k_gru1, dim3((NPIX * HD + 255) / 256), dim3(256), 0,
                     stream, gz16, gr16, inp, cor384, mot384, net, zbuf,
                     rnet16);
  conv3(w_convq, rnet16, convq_b, 128, 128, gq16, 128, nullptr, EP_NONE);
  hipLaunchKernelGGL(k_gru2, dim3((NPIX * HD + 255) / 256), dim3(256), 0,
                     stream, gq16, inp, cor384, mot384, net, zbuf, out_net,
                     net16);
  // heads (hidden buffer ch256a reused; stream-ordered so this is safe)
  conv3(w_ae1, net16, ae1_b, 256, 128, ch256a, 256, nullptr, EP_RELU);
  conv1(w_ae2, ch256a, ae2_b, 32, 256, nullptr, 0, out_ae, EP_NONE);
  conv3(w_mask1, net16, mask1_b, 256, 128, ch256a, 256, nullptr, EP_RELU);
  conv1(w_mask2, ch256a, mask2_b, 576, 256, nullptr, 0, out_mask, EP_NONE);
  conv3(w_delta1, net16, delta1_b, 256, 128, ch256a, 256, nullptr, EP_RELU);
  conv1(w_delta2, ch256a, delta2_b, 3, 256, nullptr, 0, out_delta, EP_NONE);
  conv3(w_wt1, net16, wt1_b, 256, 128, ch256a, 256, nullptr, EP_RELU);
  conv1(w_wt2, ch256a, wt2_b, 3, 256, nullptr, 0, out_wt, EP_SIG);
}
